// HyenaOperator_42915313222211
// MI455X (gfx1250) — compile-verified, matrix operand order checked by reading, not run on hardware
//
#include <hip/hip_runtime.h>
#include <hip/hip_bf16.h>
#include <math.h>

typedef __attribute__((ext_vector_type(16))) __bf16 v16bf;
typedef __attribute__((ext_vector_type(8)))  float  v8f;

#define B_   2
#define L_   8192
#define D_   512
#define FO_  64
#define DL_  (D_ * L_)                 // 4194304
#define SZ_  ((size_t)B_ * DL_)        // 8388608 floats per (B,D,L) buffer
#define MF_  8192                      // complex FFT size (real n = 16384 packed)
#define NF_  8193                      // rfft bins for n = 16384

__device__ __forceinline__ unsigned short f2bf(float f) {
  unsigned int u = __float_as_uint(f);
  u += 0x7fffu + ((u >> 16) & 1u);     // round-to-nearest-even
  return (unsigned short)(u >> 16);
}

__device__ __forceinline__ unsigned long long pack4bf(float4 v) {
  return (unsigned long long)f2bf(v.x)
       | ((unsigned long long)f2bf(v.y) << 16)
       | ((unsigned long long)f2bf(v.z) << 32)
       | ((unsigned long long)f2bf(v.w) << 48);
}

union Frag16 { v16bf v; uint4 q[2]; unsigned short u[16]; };

// ---------------------------------------------------------------------------
// Kernel 1: in_proj GEMM  proj = x @ W_in^T + b_in   (M=16384, N=1536, K=512)
// bf16 WMMA, block tile 128x128, K-step 32, 8 waves (2 WMMA-M x 4 WMMA-N each).
// LDS tiles are K-contiguous per row/col so each fragment = 2x ds_load_b128.
// Epilogue stores u/x1/x2 in transposed (B, D, L) layout (contiguous in t).
// ---------------------------------------------------------------------------
__global__ void __launch_bounds__(256)
hyena_inproj_wmma(const float* __restrict__ x, const float* __restrict__ Win,
                  const float* __restrict__ bin, float* __restrict__ ws) {
  __shared__ unsigned short As[128 * 32];   // As[row][k]
  __shared__ unsigned short Bs[128 * 32];   // Bs[col][k]  (K-contiguous!)
  const int tid  = threadIdx.x;
  const int lane = tid & 31, wave = tid >> 5;
  const int mbase = (wave >> 1) * 32;
  const int nbase = (wave & 1) * 64;
  const int j0 = blockIdx.x * 128;     // 0..1535
  const int r0 = blockIdx.y * 128;     // 0..16383

  v8f acc[2][4];
  const v8f zero = {0.f,0.f,0.f,0.f,0.f,0.f,0.f,0.f};
  for (int i = 0; i < 2; i++)
    for (int j = 0; j < 4; j++) acc[i][j] = zero;

  for (int k0 = 0; k0 < 512; k0 += 32) {
    if (k0 + 32 < 512)
      __builtin_prefetch(&x[(size_t)(r0 + (tid >> 1)) * 512 + k0 + 32], 0, 3);
    // A tile: 128 rows x 32 k of x (f32 -> bf16), K-contiguous packed stores
    for (int rep = 0; rep < 4; rep++) {
      int idx = rep * 256 + tid;
      int row = idx >> 3, c4 = idx & 7;
      const float4 v = *reinterpret_cast<const float4*>(
          &x[(size_t)(r0 + row) * 512 + k0 + c4 * 4]);
      *reinterpret_cast<unsigned long long*>(&As[row * 32 + c4 * 4]) = pack4bf(v);
    }
    // B tile: Bs[col][k] = Win[(j0+col)*512 + k0+k] -- contiguous copy
    for (int rep = 0; rep < 4; rep++) {
      int idx = rep * 256 + tid;
      int j = idx >> 3, k4 = idx & 7;
      const float4 v = *reinterpret_cast<const float4*>(
          &Win[(size_t)(j0 + j) * 512 + k0 + k4 * 4]);
      *reinterpret_cast<unsigned long long*>(&Bs[j * 32 + k4 * 4]) = pack4bf(v);
    }
    __syncthreads();

    Frag16 a[2], bf[4];
    const int Mr = lane & 15, kh8 = (lane >> 4) * 8;
    for (int mt = 0; mt < 2; mt++) {
      const int rowb = (mbase + mt * 16 + Mr) * 32;
      a[mt].q[0] = *reinterpret_cast<const uint4*>(&As[rowb + kh8]);
      a[mt].q[1] = *reinterpret_cast<const uint4*>(&As[rowb + 16 + kh8]);
    }
    for (int nt = 0; nt < 4; nt++) {
      const int colb = (nbase + nt * 16 + Mr) * 32;
      bf[nt].q[0] = *reinterpret_cast<const uint4*>(&Bs[colb + kh8]);
      bf[nt].q[1] = *reinterpret_cast<const uint4*>(&Bs[colb + 16 + kh8]);
    }
    for (int mt = 0; mt < 2; mt++)
      for (int nt = 0; nt < 4; nt++)
        acc[mt][nt] = __builtin_amdgcn_wmma_f32_16x16x32_bf16(
            false, a[mt].v, false, bf[nt].v, (short)0, acc[mt][nt], false, false);
    __syncthreads();
  }

  // Epilogue: transposed stores. C layout: lane<16 -> M=v, lane>=16 -> M=v+8
  const int nc = lane & 15, hi = lane >> 4;
  const int bblk = r0 >> 13;           // r0 / L
  const int t0 = r0 & (L_ - 1);
  for (int mt = 0; mt < 2; mt++) {
    const int tb = t0 + mbase + mt * 16 + 8 * hi;
    for (int nt = 0; nt < 4; nt++) {
      const int j = j0 + nbase + nt * 16 + nc;     // 0..1535
      const float bias = bin[j];
      const int sel = j >> 9;                       // 0:u 1:x1 2:x2
      const int jd  = j & 511;
      float* dst = ws + (size_t)sel * SZ_ + (size_t)bblk * DL_ +
                   (size_t)jd * L_ + tb;
      float4 lo = make_float4(acc[mt][nt][0] + bias, acc[mt][nt][1] + bias,
                              acc[mt][nt][2] + bias, acc[mt][nt][3] + bias);
      float4 hi4 = make_float4(acc[mt][nt][4] + bias, acc[mt][nt][5] + bias,
                               acc[mt][nt][6] + bias, acc[mt][nt][7] + bias);
      reinterpret_cast<float4*>(dst)[0] = lo;
      reinterpret_cast<float4*>(dst)[1] = hi4;
    }
  }
}

// ---------------------------------------------------------------------------
// Shared-LDS radix-2 FFT, m = 8192 complex, 256 threads, input bit-reversed.
// sign = -1 forward, +1 inverse (unnormalized).
// ---------------------------------------------------------------------------
__device__ void fft_lds(float* re, float* im, int tid, float sign) {
  int stage = 0;
  for (int len = 2; len <= MF_; len <<= 1) {
    const int half = len >> 1;
    const float w = sign * 3.14159265358979323846f / (float)half;
    for (int idx = tid; idx < (MF_ >> 1); idx += 256) {
      const int pos = idx & (half - 1);
      const int i0 = ((idx >> stage) << (stage + 1)) + pos;
      const int i1 = i0 + half;
      float s, c;
      __sincosf(w * (float)pos, &s, &c);
      const float xr = re[i1], xi = im[i1];
      const float tr = c * xr - s * xi;
      const float ti = c * xi + s * xr;
      const float ur = re[i0], ui = im[i0];
      re[i0] = ur + tr; im[i0] = ui + ti;
      re[i1] = ur - tr; im[i1] = ui - ti;
    }
    __syncthreads();
    stage++;
  }
}

// genomic positional encoding value at (t, f), FO = 64
__device__ float genomic_pe(int t, int f) {
  const float TWO_PI = 6.28318530717958647692f;
  const int j = f >> 1;
  const float div = __expf(-9.21034037197618f * (float)(2 * j) / 64.0f);
  const float arg = (float)t * div;
  float v = (f & 1) ? __cosf(arg) : __sinf(arg);
  if (f == 60) v += 0.1f  * __sinf(TWO_PI * (float)(t % 3) / 3.0f);
  if (f == 61) v += 0.1f  * __sinf(TWO_PI * (float)t / 6.0f);
  if (f == 55) v += 0.05f * __sinf(TWO_PI * (float)t / 8.0f);
  if (f == 53) v += 0.05f * __sinf(TWO_PI * (float)t / 10.0f);
  if (f == 42) v += 0.05f * __sinf(TWO_PI * (float)t / 21.0f);
  return v;
}

__device__ float fc_eval(int t, int d,
                         const float* __restrict__ fr, const float* __restrict__ ph,
                         const float* __restrict__ am, const float* __restrict__ de,
                         const float* __restrict__ fW, float fb) {
  float acc = fb;
  const float tf = (float)t;
  const float* fwrow = fW + (size_t)d * 64;
  for (int g = 0; g < 32; g++) {
    const float env = am[g] * __expf(de[g] * tf);
    float s, c;
    __sincosf(fr[g] * tf + ph[g], &s, &c);
    acc += fwrow[g]      * (env * s + genomic_pe(t, g));
    acc += fwrow[g + 32] * (env * c + genomic_pe(t, g + 32));
  }
  return acc;
}

// ---------------------------------------------------------------------------
// Kernel 2: implicit filter synthesis + rfft -> Kf[d][0..8192] complex
// One block per channel d. Real 16384-pt rfft via packed 8192-pt complex FFT.
// ---------------------------------------------------------------------------
__global__ void __launch_bounds__(256)
hyena_filter_fft(const float* __restrict__ fr, const float* __restrict__ ph,
                 const float* __restrict__ am, const float* __restrict__ de,
                 const float* __restrict__ fW, const float* __restrict__ fb,
                 float* __restrict__ Kf) {
  __shared__ float sre[MF_];
  __shared__ float sim[MF_];
  const int d = blockIdx.x, tid = threadIdx.x;
  const float fbd = fb[d];

  for (int k = tid; k < MF_; k += 256) {
    const int rk = __brev((unsigned)k) >> 19;   // 13-bit reverse
    float zr = 0.f, zi = 0.f;
    if (k < 4096) {                             // fc has length L = 8192
      zr = fc_eval(2 * k,     d, fr, ph, am, de, fW, fbd);
      zi = fc_eval(2 * k + 1, d, fr, ph, am, de, fW, fbd);
    }
    sre[rk] = zr; sim[rk] = zi;
  }
  __syncthreads();
  fft_lds(sre, sim, tid, -1.f);

  float* kfd = Kf + (size_t)d * NF_ * 2;
  for (int k = tid; k <= 4096; k += 256) {
    if (k == 0) {
      const float e0 = sre[0], o0 = sim[0];
      kfd[0] = e0 + o0;           kfd[1] = 0.f;            // X[0]
      kfd[2 * 8192] = e0 - o0;    kfd[2 * 8192 + 1] = 0.f; // X[m]
    } else {
      const float ar = sre[k], ai = sim[k];
      const float br = sre[8192 - k], bi = sim[8192 - k];
      const float Er = 0.5f * (ar + br), Ei = 0.5f * (ai - bi);
      const float Or = 0.5f * (ai + bi), Oi = 0.5f * (br - ar);
      float c, s;
      __sincosf(-3.14159265358979324f * (float)k / 8192.0f, &s, &c);
      const float WOr = c * Or - s * Oi, WOi = c * Oi + s * Or;
      kfd[2 * k]     = Er + WOr;                 // X[k] = E + W*O
      kfd[2 * k + 1] = Ei + WOi;
      kfd[2 * (8192 - k)]     = Er - WOr;        // X[m-k] = conj(E - W*O)
      kfd[2 * (8192 - k) + 1] = -(Ei - WOi);
    }
  }
}

// ---------------------------------------------------------------------------
// Kernel 3: depthwise causal conv (K=3) + segment mask, on transposed layout.
// ---------------------------------------------------------------------------
__global__ void __launch_bounds__(256)
hyena_convmask(const int* __restrict__ seg, const float* __restrict__ cw,
               const float* __restrict__ cb, float* __restrict__ ws) {
  const int b = blockIdx.x >> 9, d = blockIdx.x & 511;
  const int tid = threadIdx.x;
  const float w0 = cw[d * 3 + 0], w1 = cw[d * 3 + 1], w2 = cw[d * 3 + 2];
  const float bias = cb[d];
  const float* u = ws + (size_t)b * DL_ + (size_t)d * L_;           // uT
  float* ucp = ws + 3 * SZ_ + (size_t)b * DL_ + (size_t)d * L_;     // ucT
  const int* bp = seg + (size_t)b * L_;
  for (int t = tid; t < L_; t += 256) {
    const float um2 = (t >= 2) ? u[t - 2] : 0.f;
    const float um1 = (t >= 1) ? u[t - 1] : 0.f;
    const float v = w0 * um2 + w1 * um1 + w2 * u[t] + bias;
    const int cnt = bp[t] + ((t >= 1) ? bp[t - 1] : 0) +
                    ((t + 1 < L_) ? bp[t + 1] : 0);
    ucp[t] = v * exp2f(-(float)cnt);
  }
}

// ---------------------------------------------------------------------------
// Kernel 4: per-(b,d) FFT causal convolution of x1 with Kf, fused gating.
// x1 column is staged global->LDS with ASYNC loads (CDNA5 ASYNCcnt path).
// g = x2 * x_filt * u_conv * sigmoid(u), written over x1T.
// ---------------------------------------------------------------------------
__global__ void __launch_bounds__(256)
hyena_fftconv_gate(const float* __restrict__ Kf, float* __restrict__ ws) {
  __shared__ float sre[MF_];
  __shared__ float sim[MF_];
  __shared__ float stg[L_];                   // 32 KB async staging for x1
  const int tid = threadIdx.x;
  const int b = blockIdx.x >> 9, d = blockIdx.x & 511;
  const size_t base = (size_t)b * DL_ + (size_t)d * L_;
  const float* x1 = ws + SZ_ + base;
  const float* x2 = ws + 2 * SZ_ + base;
  const float* uu = ws + base;
  const float* uc = ws + 3 * SZ_ + base;
  float* g = ws + SZ_ + base;                 // overwrite x1T (fully consumed)
  const float* kfd = Kf + (size_t)d * NF_ * 2;

  // Async global->LDS copy of the whole x1 column (8192 f32 = 32 KB).
  {
    for (int it = 0; it < 8; it++) {
      const int off = (it * 256 + tid) * 4;   // float index, 16 B per lane
      unsigned laddr = (unsigned)(size_t)(&stg[off]);
      unsigned long long gaddr = (unsigned long long)(size_t)(x1 + off);
      asm volatile("global_load_async_to_lds_b128 %0, %1, off"
                   :: "v"(laddr), "v"(gaddr) : "memory");
    }
    asm volatile("s_wait_asynccnt 0x0" ::: "memory");
  }
  __syncthreads();

  for (int k = tid; k < MF_; k += 256) {
    const int rk = __brev((unsigned)k) >> 19;
    float zr = 0.f, zi = 0.f;
    if (k < 4096) { zr = stg[2 * k]; zi = stg[2 * k + 1]; }  // zero-pad to 2L
    sre[rk] = zr; sim[rk] = zi;
  }
  __syncthreads();
  fft_lds(sre, sim, tid, -1.f);

  const float sc = 1.0f / 8192.0f;            // 1/m (pack-route irfft scale)
  for (int k = tid; k <= 4096; k += 256) {
    if (k == 0) {
      const float e0 = sre[0], o0 = sim[0];
      const float X0 = e0 + o0, Xm = e0 - o0;
      const float y0r = X0 * kfd[0] * sc,            y0i = X0 * kfd[1] * sc;
      const float ymr = Xm * kfd[2 * 8192] * sc,     ymi = Xm * kfd[2 * 8192 + 1] * sc;
      const float Epr = 0.5f * (y0r + ymr), Epi = 0.5f * (y0i - ymi);
      const float Opr = 0.5f * (y0r - ymr), Opi = 0.5f * (y0i + ymi);
      sre[0] = Epr - Opi; sim[0] = Epi + Opr;
    } else {
      const float ar = sre[k], ai = sim[k];
      const float br = sre[8192 - k], bi = sim[8192 - k];
      const float Er = 0.5f * (ar + br), Ei = 0.5f * (ai - bi);
      const float Or = 0.5f * (ai + bi), Oi = 0.5f * (br - ar);
      float c, s;
      __sincosf(-3.14159265358979324f * (float)k / 8192.0f, &s, &c);
      const float WOr = c * Or - s * Oi, WOi = c * Oi + s * Or;
      const float Pr = Er + WOr, Pi = Ei + WOi;       // X[k]
      const float Qr = Er - WOr, Qi = -(Ei - WOi);    // X[m-k]
      const float k1r = kfd[2 * k], k1i = kfd[2 * k + 1];
      const float k2r = kfd[2 * (8192 - k)], k2i = kfd[2 * (8192 - k) + 1];
      const float Ykr = (Pr * k1r - Pi * k1i) * sc;
      const float Yki = (Pr * k1i + Pi * k1r) * sc;
      const float Ymr = (Qr * k2r - Qi * k2i) * sc;
      const float Ymi = (Qr * k2i + Qi * k2r) * sc;
      const float Epr = 0.5f * (Ykr + Ymr), Epi = 0.5f * (Yki - Ymi);
      const float Tr  = 0.5f * (Ykr - Ymr), Ti  = 0.5f * (Yki + Ymi);
      const float Opr = c * Tr + s * Ti;              // conj(W) * T
      const float Opi = c * Ti - s * Tr;
      sre[k] = Epr - Opi; sim[k] = Epi + Opr;         // Z'[k]
      const float Wtr = c * Tr + s * Ti;              // W * conj(T)
      const float Wti = s * Tr - c * Ti;
      sre[8192 - k] = Epr - Wti;                      // Z'[m-k]
      sim[8192 - k] = -Epi + Wtr;
    }
  }
  __syncthreads();
  for (int i = tid; i < MF_; i += 256) {              // bit-reverse for inverse
    const int j = __brev((unsigned)i) >> 19;
    if (j > i) {
      const float tr = sre[i]; sre[i] = sre[j]; sre[j] = tr;
      const float ti = sim[i]; sim[i] = sim[j]; sim[j] = ti;
    }
  }
  __syncthreads();
  fft_lds(sre, sim, tid, 1.f);

  for (int t = tid; t < L_; t += 256) {
    const float xf = (t & 1) ? sim[t >> 1] : sre[t >> 1];
    const float uv = uu[t];
    const float sg = 1.0f / (1.0f + __expf(-uv));
    g[t] = x2[t] * xf * uc[t] * sg;
  }
}

// ---------------------------------------------------------------------------
// Kernel 5: out_proj GEMM with transposed-A + bias + residual (M=16384,N=512,K=512)
// ---------------------------------------------------------------------------
__global__ void __launch_bounds__(256)
hyena_outproj_wmma(const float* __restrict__ gT, const float* __restrict__ Wout,
                   const float* __restrict__ bout, const float* __restrict__ x,
                   float* __restrict__ h) {
  __shared__ unsigned short As[128 * 32];   // As[t][k]
  __shared__ unsigned short Bs[128 * 32];   // Bs[col][k]
  const int tid  = threadIdx.x;
  const int lane = tid & 31, wave = tid >> 5;
  const int mbase = (wave >> 1) * 32;
  const int nbase = (wave & 1) * 64;
  const int j0 = blockIdx.x * 128;
  const int r0 = blockIdx.y * 128;
  const int bblk = r0 >> 13;
  const int t0 = r0 & (L_ - 1);

  v8f acc[2][4];
  const v8f zero = {0.f,0.f,0.f,0.f,0.f,0.f,0.f,0.f};
  for (int i = 0; i < 2; i++)
    for (int j = 0; j < 4; j++) acc[i][j] = zero;

  for (int k0 = 0; k0 < 512; k0 += 32) {
    // A tile from gT: A[t][k] = gT[b*DL + (k0+k)*L + t0+t]  (contiguous in t)
    for (int rep = 0; rep < 4; rep++) {
      int idx = rep * 256 + tid;
      int kk = idx >> 5, t4 = idx & 31;
      const float4 v = *reinterpret_cast<const float4*>(
          &gT[(size_t)bblk * DL_ + (size_t)(k0 + kk) * L_ + t0 + t4 * 4]);
      As[(t4 * 4 + 0) * 32 + kk] = f2bf(v.x);
      As[(t4 * 4 + 1) * 32 + kk] = f2bf(v.y);
      As[(t4 * 4 + 2) * 32 + kk] = f2bf(v.z);
      As[(t4 * 4 + 3) * 32 + kk] = f2bf(v.w);
    }
    // B tile: Bs[col][k] = Wout[(j0+col)*512 + k0+k] -- contiguous copy
    for (int rep = 0; rep < 4; rep++) {
      int idx = rep * 256 + tid;
      int j = idx >> 3, k4 = idx & 7;
      const float4 v = *reinterpret_cast<const float4*>(
          &Wout[(size_t)(j0 + j) * 512 + k0 + k4 * 4]);
      *reinterpret_cast<unsigned long long*>(&Bs[j * 32 + k4 * 4]) = pack4bf(v);
    }
    __syncthreads();

    Frag16 a[2], bf[4];
    const int Mr = lane & 15, kh8 = (lane >> 4) * 8;
    for (int mt = 0; mt < 2; mt++) {
      const int rowb = (mbase + mt * 16 + Mr) * 32;
      a[mt].q[0] = *reinterpret_cast<const uint4*>(&As[rowb + kh8]);
      a[mt].q[1] = *reinterpret_cast<const uint4*>(&As[rowb + 16 + kh8]);
    }
    for (int nt = 0; nt < 4; nt++) {
      const int colb = (nbase + nt * 16 + Mr) * 32;
      bf[nt].q[0] = *reinterpret_cast<const uint4*>(&Bs[colb + kh8]);
      bf[nt].q[1] = *reinterpret_cast<const uint4*>(&Bs[colb + 16 + kh8]);
    }
    for (int mt = 0; mt < 2; mt++)
      for (int nt = 0; nt < 4; nt++)
        acc[mt][nt] = __builtin_amdgcn_wmma_f32_16x16x32_bf16(
            false, a[mt].v, false, bf[nt].v, (short)0, acc[mt][nt], false, false);
    __syncthreads();
  }

  const int nc = lane & 15, hi = lane >> 4;
  for (int mt = 0; mt < 2; mt++) {
    for (int nt = 0; nt < 4; nt++) {
      const int j = j0 + nbase + nt * 16 + nc;
      const float bias = bout[j];
      for (int v = 0; v < 8; v++) {
        const int r = r0 + mbase + mt * 16 + 8 * hi + v;
        const size_t off = (size_t)r * 512 + j;
        h[off] = acc[mt][nt][v] + bias + x[off];
      }
    }
  }
}

// ---------------------------------------------------------------------------
// Kernel 6: LayerNorm over D=512 per row.
// ---------------------------------------------------------------------------
__global__ void __launch_bounds__(256)
hyena_layernorm(const float* __restrict__ h, const float* __restrict__ gam,
                const float* __restrict__ bet, float* __restrict__ out) {
  __shared__ float s1[256];
  __shared__ float s2[256];
  const int r = blockIdx.x, tid = threadIdx.x;
  const float* hr = h + (size_t)r * 512;
  const float v0 = hr[tid], v1 = hr[tid + 256];
  s1[tid] = v0 + v1;
  s2[tid] = v0 * v0 + v1 * v1;
  __syncthreads();
  for (int off = 128; off > 0; off >>= 1) {
    if (tid < off) { s1[tid] += s1[tid + off]; s2[tid] += s2[tid + off]; }
    __syncthreads();
  }
  const float mu  = s1[0] * (1.0f / 512.0f);
  const float var = s2[0] * (1.0f / 512.0f) - mu * mu;
  const float inv = rsqrtf(var + 1e-5f);
  out[(size_t)r * 512 + tid]       = (v0 - mu) * inv * gam[tid]       + bet[tid];
  out[(size_t)r * 512 + tid + 256] = (v1 - mu) * inv * gam[tid + 256] + bet[tid + 256];
}

// ---------------------------------------------------------------------------
extern "C" void kernel_launch(void* const* d_in, const int* in_sizes, int n_in,
                              void* d_out, int out_size, void* d_ws, size_t ws_size,
                              hipStream_t stream) {
  (void)in_sizes; (void)n_in; (void)out_size; (void)ws_size;
  const float* x    = (const float*)d_in[0];
  const int*   seg  = (const int*)d_in[1];
  const float* Win  = (const float*)d_in[2];
  const float* bin  = (const float*)d_in[3];
  const float* cw   = (const float*)d_in[4];
  const float* cb   = (const float*)d_in[5];
  const float* fr   = (const float*)d_in[6];
  const float* ph   = (const float*)d_in[7];
  const float* am   = (const float*)d_in[8];
  const float* de   = (const float*)d_in[9];
  const float* fW   = (const float*)d_in[10];
  const float* fb   = (const float*)d_in[11];
  const float* Wout = (const float*)d_in[12];
  const float* bout = (const float*)d_in[13];
  const float* lng  = (const float*)d_in[14];
  const float* lnb  = (const float*)d_in[15];

  float* ws = (float*)d_ws;
  // ws layout (floats): [0,SZ) uT (reused as h), [SZ,2SZ) x1T (reused as gT),
  // [2SZ,3SZ) x2T, [3SZ,4SZ) ucT, [4SZ, 4SZ + 512*8193*2) Kf
  float* Kf = ws + 4 * SZ_;
  float* h  = ws;            // reuse uT region after kernel 4 consumed u

  hyena_inproj_wmma<<<dim3(12, 128), dim3(256), 0, stream>>>(x, Win, bin, ws);
  hyena_filter_fft<<<dim3(512), dim3(256), 0, stream>>>(fr, ph, am, de, fW, fb, Kf);
  hyena_convmask<<<dim3(1024), dim3(256), 0, stream>>>(seg, cw, cb, ws);
  hyena_fftconv_gate<<<dim3(1024), dim3(256), 0, stream>>>(Kf, ws);
  hyena_outproj_wmma<<<dim3(4, 128), dim3(256), 0, stream>>>(ws + SZ_, Wout, bout, x, h);
  hyena_layernorm<<<dim3(16384), dim3(256), 0, stream>>>(h, lng, lnb, (float*)d_out);
}
